// ResidualQuantizer_39977555591198
// MI455X (gfx1250) — matrix operand order repaired and checked by reading, not compiled, not run on hardware
//
#include <hip/hip_runtime.h>
#include <hip/hip_bf16.h>

// ---------------------------------------------------------------------------
// VQ-VAE residual quantizer fused for gfx1250 (MI455X, wave32, WMMA).
//   qi = x*qw^T + qb ; idx = argmin_k |qi - emb_k|^2 ; q = emb[idx]
//   out = x - (q*pw^T + pb) ; loss = 1.25 * mean((q - qi)^2)
// All three GEMMs run on v_wmma_f32_16x16x32_bf16 with split-bf16 (hi+lo)
// operands for ~fp32 accuracy: A*B ~= Ah*Bh + Ah*Bl + Al*Bh.
// ---------------------------------------------------------------------------

typedef __attribute__((ext_vector_type(16))) __bf16 v16bf;
typedef __attribute__((ext_vector_type(8)))  float  v8f;

#define TM   64      // pixels per workgroup tile
#define LDK  66      // padded k-stride (ushorts) for LDS tiles
#define ZCH  256     // input channels (Z)
#define DCH  64      // latent dim (D)
#define KCB  1024    // codebook size
#define HW   4096    // H*W
#define NPIX_TOT 131072         // B*H*W
#define NWG  2048               // NPIX_TOT / TM
#define LOSS_SCALE (1.25f / 8388608.f)   // 1.25 / (B*D*H*W)

__device__ __forceinline__ unsigned short f2bf(float x) {
  union { float f; unsigned u; } a; a.f = x;
  return (unsigned short)(a.u >> 16);
}
__device__ __forceinline__ float bf2f(unsigned short h) {
  union { float f; unsigned u; } a; a.u = ((unsigned)h) << 16;
  return a.f;
}
__device__ __forceinline__ void split2(float x, unsigned short& h, unsigned short& l) {
  h = f2bf(x);
  l = f2bf(x - bf2f(h));
}

union Frag { v16bf v; unsigned u[8]; };

// A fragment (16x32 bf16, ISA layout): lane<16 -> M=lane, k pairs 0..7 then 16..23;
// lane>=16 -> M=lane-16, k pairs 8..15 then 24..31.  LDS tile is [row][k], k-major.
__device__ __forceinline__ v16bf lds_load_a(const unsigned short* A, int row0, int kbase) {
  int lane = threadIdx.x & 31;
  int m = row0 + (lane & 15);
  int koff = (lane & 16) ? 8 : 0;
  const unsigned short* rp = A + m * LDK + kbase;
  Frag f;
#pragma unroll
  for (int g = 0; g < 8; ++g) {
    int k0 = 2 * g + ((g & 4) ? 8 : 0) + koff;
    f.u[g] = *(const unsigned*)(rp + k0);
  }
  return f.v;
}

// B fragment (32x16 bf16): lane<16 -> N=lane, k 0..7 then 16..23; lane>=16 -> N=lane-16, k 8..15 then 24..31.
// LDS tile is [n][k], k-major (matches emb/qw/pw natural layout).
__device__ __forceinline__ v16bf lds_load_b(const unsigned short* B, int row0, int kbase) {
  int lane = threadIdx.x & 31;
  int n = row0 + (lane & 15);
  int koff = (lane & 16) ? 8 : 0;
  const unsigned short* rp = B + n * LDK + kbase + koff;
  Frag f;
#pragma unroll
  for (int g = 0; g < 8; ++g) f.u[g] = *(const unsigned*)(rp + 2 * g + ((g & 4) ? 8 : 0));
  return f.v;
}

__device__ __forceinline__ void wmma3(v8f& acc, v16bf ah, v16bf al, v16bf bh, v16bf bl) {
  acc = __builtin_amdgcn_wmma_f32_16x16x32_bf16(false, ah, false, bh, (short)0, acc, false, false);
  acc = __builtin_amdgcn_wmma_f32_16x16x32_bf16(false, ah, false, bl, (short)0, acc, false, false);
  acc = __builtin_amdgcn_wmma_f32_16x16x32_bf16(false, al, false, bh, (short)0, acc, false, false);
}

// ---------------------------------------------------------------------------
// Prep: split emb/qw/pw into bf16 hi/lo, precompute |emb_k|^2.
// ---------------------------------------------------------------------------
__global__ void vq_prep(const float* __restrict__ emb, const float* __restrict__ qw,
                        const float* __restrict__ pw,
                        unsigned short* __restrict__ emb_hi, unsigned short* __restrict__ emb_lo,
                        unsigned short* __restrict__ qw_hi,  unsigned short* __restrict__ qw_lo,
                        unsigned short* __restrict__ pw_hi,  unsigned short* __restrict__ pw_lo,
                        float* __restrict__ e2) {
  int t = blockIdx.x * blockDim.x + threadIdx.x;
  if (t < KCB * DCH) { unsigned short h, l; split2(emb[t], h, l); emb_hi[t] = h; emb_lo[t] = l; }
  if (t < DCH * ZCH) { unsigned short h, l; split2(qw[t], h, l);  qw_hi[t] = h;  qw_lo[t] = l; }
  if (t < ZCH * DCH) { unsigned short h, l; split2(pw[t], h, l);  pw_hi[t] = h;  pw_lo[t] = l; }
  if (t < KCB) {
    float s = 0.f;
    const float* row = emb + t * DCH;
#pragma unroll 8
    for (int d = 0; d < DCH; ++d) s += row[d] * row[d];
    e2[t] = s;
  }
}

// ---------------------------------------------------------------------------
// Fused main kernel: 128 threads (4 waves), 64-pixel tile per workgroup.
// ---------------------------------------------------------------------------
__global__ __launch_bounds__(128) void vq_fused(
    const float* __restrict__ x,  const float* __restrict__ qb,
    const float* __restrict__ pb, const float* __restrict__ emb,
    const unsigned short* __restrict__ emb_hi, const unsigned short* __restrict__ emb_lo,
    const unsigned short* __restrict__ qw_hi,  const unsigned short* __restrict__ qw_lo,
    const unsigned short* __restrict__ pw_hi,  const unsigned short* __restrict__ pw_lo,
    const float* __restrict__ e2, float* __restrict__ out, float* __restrict__ partials) {
  __shared__ unsigned short sA_hi[TM * LDK], sA_lo[TM * LDK];   // x chunk, later q
  __shared__ unsigned short sF_hi[TM * LDK], sF_lo[TM * LDK];   // qi (split)
  __shared__ unsigned short sB_hi[64 * LDK], sB_lo[64 * LDK];   // qw / emb / pw chunks
  __shared__ float sE2[64];
  __shared__ int   sIdx[TM];
  __shared__ float sPost[16 * TM];
  __shared__ float sRed[4];

  const int tid  = threadIdx.x;
  const int lane = tid & 31;
  const int wave = tid >> 5;
  const int wg   = blockIdx.x;
  const int b    = wg >> 6;              // 64 tiles per image
  const int hw0  = (wg & 63) * TM;
  const int row0 = wave * 16;
  const int moff = (lane & 16) ? 8 : 0;  // C-matrix M offset for upper half-wave
  const int ncol = lane & 15;            // C-matrix N column

  const float* xb = x + (size_t)b * ZCH * HW + hw0;
  const v8f vzero = {0.f, 0.f, 0.f, 0.f, 0.f, 0.f, 0.f, 0.f};

  // ---------------- Stage 1: qi = x * qw^T + qb (M=64, N=64, K=256) ----------
  v8f acc[4];
#pragma unroll
  for (int i = 0; i < 4; ++i) acc[i] = vzero;

  for (int kc = 0; kc < 4; ++kc) {            // K chunks of 64 channels
#pragma unroll
    for (int i = 0; i < 32; ++i) {            // x[c][p] -> sA[p][c] (split bf16)
      int idx = tid + i * 128;
      int c = idx >> 6, p = idx & 63;
      float v = xb[(size_t)(kc * 64 + c) * HW + p];
      unsigned short h, l; split2(v, h, l);
      sA_hi[p * LDK + c] = h; sA_lo[p * LDK + c] = l;
    }
#pragma unroll
    for (int i = 0; i < 32; ++i) {            // qw[d][c] chunk -> sB[d][c]
      int idx = tid + i * 128;
      int d = idx >> 6, c = idx & 63;
      sB_hi[d * LDK + c] = qw_hi[d * ZCH + kc * 64 + c];
      sB_lo[d * LDK + c] = qw_lo[d * ZCH + kc * 64 + c];
    }
    __syncthreads();
#pragma unroll
    for (int ks = 0; ks < 2; ++ks) {
      v16bf ah = lds_load_a(sA_hi, row0, ks * 32);
      v16bf al = lds_load_a(sA_lo, row0, ks * 32);
#pragma unroll
      for (int nt = 0; nt < 4; ++nt) {
        v16bf bh = lds_load_b(sB_hi, nt * 16, ks * 32);
        v16bf bl = lds_load_b(sB_lo, nt * 16, ks * 32);
        wmma3(acc[nt], ah, al, bh, bl);
      }
    }
    __syncthreads();
  }
  // finalize qi, split to sF
#pragma unroll
  for (int nt = 0; nt < 4; ++nt) {
    int d = nt * 16 + ncol;
    float qbv = qb[d];
#pragma unroll
    for (int r = 0; r < 8; ++r) {
      int m = row0 + moff + r;
      float v = acc[nt][r] + qbv;
      unsigned short h, l; split2(v, h, l);
      sF_hi[m * LDK + d] = h; sF_lo[m * LDK + d] = l;
    }
  }
  __syncthreads();

  // ---------------- Stage 2: argmin_k ( |e_k|^2 - 2 f.e_k ) ------------------
  float bestS[8]; int bestI[8];
#pragma unroll
  for (int r = 0; r < 8; ++r) { bestS[r] = 3.4e38f; bestI[r] = 0; }

  for (int cc = 0; cc < 16; ++cc) {           // codebook chunks of 64
#pragma unroll
    for (int i = 0; i < 32; ++i) {
      int idx = tid + i * 128;
      int n = idx >> 6, k = idx & 63;
      sB_hi[n * LDK + k] = emb_hi[(cc * 64 + n) * DCH + k];
      sB_lo[n * LDK + k] = emb_lo[(cc * 64 + n) * DCH + k];
    }
    if (tid < 64) sE2[tid] = e2[cc * 64 + tid];
    __syncthreads();

    // prefetch next codebook chunk into cache while WMMAs run on this one
    if (cc < 15) {
      // each thread prefetches one 128B line of the next chunk (hi or lo array)
      const unsigned short* nph = emb_hi + (cc + 1) * 64 * DCH;
      const unsigned short* npl = emb_lo + (cc + 1) * 64 * DCH;
      if (tid < 64) __builtin_prefetch(nph + tid * 64, 0, 1);
      else          __builtin_prefetch(npl + (tid - 64) * 64, 0, 1);
    }

    v8f dacc[4];
#pragma unroll
    for (int i = 0; i < 4; ++i) dacc[i] = vzero;
#pragma unroll
    for (int ks = 0; ks < 2; ++ks) {
      v16bf ah = lds_load_a(sF_hi, row0, ks * 32);
      v16bf al = lds_load_a(sF_lo, row0, ks * 32);
#pragma unroll
      for (int nt = 0; nt < 4; ++nt) {
        v16bf bh = lds_load_b(sB_hi, nt * 16, ks * 32);
        v16bf bl = lds_load_b(sB_lo, nt * 16, ks * 32);
        wmma3(dacc[nt], ah, al, bh, bl);
      }
    }
#pragma unroll
    for (int nt = 0; nt < 4; ++nt) {
      int nl = nt * 16 + ncol;
      float ev = sE2[nl];
      int code = cc * 64 + nl;
#pragma unroll
      for (int r = 0; r < 8; ++r) {
        float sc = ev - 2.0f * dacc[nt][r];
        if (sc < bestS[r]) { bestS[r] = sc; bestI[r] = code; }
      }
    }
    __syncthreads();
  }
  // cross-lane argmin over the 16 N columns (first-min tie-break on index)
#pragma unroll
  for (int r = 0; r < 8; ++r) {
#pragma unroll
    for (int mask = 1; mask <= 8; mask <<= 1) {
      float os = __shfl_xor(bestS[r], mask, 32);
      int   oi = __shfl_xor(bestI[r], mask, 32);
      if (os < bestS[r] || (os == bestS[r] && oi < bestI[r])) { bestS[r] = os; bestI[r] = oi; }
    }
    if (ncol == 0) sIdx[row0 + moff + r] = bestI[r];
  }
  __syncthreads();

  // ---------------- Stage 3a: gather q, loss partial -------------------------
  float lsum = 0.f;
#pragma unroll
  for (int i = 0; i < 32; ++i) {
    int idx = tid + i * 128;
    int m = idx >> 6, d = idx & 63;
    int code = sIdx[m];
    float qv = emb[code * DCH + d];
    float qiv = bf2f(sF_hi[m * LDK + d]) + bf2f(sF_lo[m * LDK + d]);
    float diff = qv - qiv;
    lsum += diff * diff;
    unsigned short h, l; split2(qv, h, l);
    sA_hi[m * LDK + d] = h; sA_lo[m * LDK + d] = l;   // sA now holds q
  }
#pragma unroll
  for (int mask = 16; mask; mask >>= 1) lsum += __shfl_xor(lsum, mask, 32);
  if (lane == 0) sRed[wave] = lsum;
  __syncthreads();
  if (tid == 0) partials[wg] = sRed[0] + sRed[1] + sRed[2] + sRed[3];
  __syncthreads();

  // ---------------- Stage 3b: out = x - (q * pw^T + pb) ----------------------
  for (int zc = 0; zc < 4; ++zc) {            // output-channel chunks of 64
#pragma unroll
    for (int i = 0; i < 32; ++i) {
      int idx = tid + i * 128;
      int z = idx >> 6, d = idx & 63;
      sB_hi[z * LDK + d] = pw_hi[(zc * 64 + z) * DCH + d];
      sB_lo[z * LDK + d] = pw_lo[(zc * 64 + z) * DCH + d];
    }
    __syncthreads();
#pragma unroll
    for (int nt = 0; nt < 4; ++nt) {
      v8f pacc = vzero;
#pragma unroll
      for (int ks = 0; ks < 2; ++ks) {
        v16bf ah = lds_load_a(sA_hi, row0, ks * 32);
        v16bf al = lds_load_a(sA_lo, row0, ks * 32);
        v16bf bh = lds_load_b(sB_hi, nt * 16, ks * 32);
        v16bf bl = lds_load_b(sB_lo, nt * 16, ks * 32);
        wmma3(pacc, ah, al, bh, bl);
      }
#pragma unroll
      for (int r = 0; r < 8; ++r) {
        int m = row0 + moff + r;
        sPost[ncol * TM + m] = pacc[r];
      }
      __syncthreads();
#pragma unroll
      for (int i = 0; i < 8; ++i) {           // 16 z-rows x 64 px, coalesced
        int idx = tid + i * 128;
        int zl = idx >> 6, p = idx & 63;
        int z = zc * 64 + nt * 16 + zl;
        size_t o = (size_t)b * ZCH * HW + (size_t)z * HW + hw0 + p;
        out[o] = x[o] - (sPost[zl * TM + p] + pb[z]);
      }
      __syncthreads();
    }
  }
}

// ---------------------------------------------------------------------------
// Deterministic fixed-order loss reduction.
// ---------------------------------------------------------------------------
__global__ void vq_reduce_loss(const float* __restrict__ partials, float* __restrict__ loss_out) {
  __shared__ float red[256];
  float s = 0.f;
  for (int i = threadIdx.x; i < NWG; i += 256) s += partials[i];
  red[threadIdx.x] = s;
  __syncthreads();
  for (int off = 128; off; off >>= 1) {
    if (threadIdx.x < off) red[threadIdx.x] += red[threadIdx.x + off];
    __syncthreads();
  }
  if (threadIdx.x == 0) loss_out[0] = red[0] * LOSS_SCALE;
}

extern "C" void kernel_launch(void* const* d_in, const int* in_sizes, int n_in,
                              void* d_out, int out_size, void* d_ws, size_t ws_size,
                              hipStream_t stream) {
  const float* x   = (const float*)d_in[0];
  const float* qw  = (const float*)d_in[1];
  const float* qb  = (const float*)d_in[2];
  const float* emb = (const float*)d_in[3];
  const float* pw  = (const float*)d_in[4];
  const float* pb  = (const float*)d_in[5];
  float* out = (float*)d_out;

  char* ws = (char*)d_ws;
  unsigned short* emb_hi = (unsigned short*)(ws + 0);
  unsigned short* emb_lo = (unsigned short*)(ws + 131072);
  unsigned short* qw_hi  = (unsigned short*)(ws + 262144);
  unsigned short* qw_lo  = (unsigned short*)(ws + 294912);
  unsigned short* pw_hi  = (unsigned short*)(ws + 327680);
  unsigned short* pw_lo  = (unsigned short*)(ws + 360448);
  float* e2       = (float*)(ws + 393216);
  float* partials = (float*)(ws + 397312);

  vq_prep<<<256, 256, 0, stream>>>(emb, qw, pw, emb_hi, emb_lo, qw_hi, qw_lo,
                                   pw_hi, pw_lo, e2);
  vq_fused<<<NWG, 128, 0, stream>>>(x, qb, pb, emb, emb_hi, emb_lo, qw_hi, qw_lo,
                                    pw_hi, pw_lo, e2, out, partials);
  vq_reduce_loss<<<1, 256, 0, stream>>>(partials, out + 33554432);
}
